// Flux2Attention_18098992185470
// MI455X (gfx1250) — matrix operand order repaired and checked by reading, not compiled, not run on hardware
//
#include <hip/hip_runtime.h>
#include <hip/hip_bf16.h>

// ---------------------------------------------------------------------------
// Flux2 structured attention block for MI455X (gfx1250, wave32, WMMA).
// Pipeline: 6x GEMM(bias) -> RMSNorm+RoPE+bf16 cvt -> 3x flash-attention
// launches (txt->all, img->all, ref->ref) -> 2x output GEMM(bias).
// All matmuls use v_wmma_f32_16x16x32_bf16 with f32 accumulation.
// Attention K-tiles are staged with global_load_async_to_lds_b128 (ASYNCcnt).
// ---------------------------------------------------------------------------

typedef __attribute__((ext_vector_type(4)))  __bf16 bf16x4;
typedef __attribute__((ext_vector_type(8)))  __bf16 bf16x8;
typedef __attribute__((ext_vector_type(16))) __bf16 bf16x16;
typedef __attribute__((ext_vector_type(8)))  float  f32x8;

#define SHUF16(lo, hi) __builtin_shufflevector((lo), (hi), 0,1,2,3,4,5,6,7,8,9,10,11,12,13,14,15)

// Native fptrunc -> v_cvt_pk_bf16_f32 (RNE), instead of software rounding.
__device__ __forceinline__ __bf16 f2bf(float f) { return (__bf16)f; }

__device__ __forceinline__ f32x8 wmma_bf16(bf16x16 a, bf16x16 b, f32x8 c) {
  return __builtin_amdgcn_wmma_f32_16x16x32_bf16(false, a, false, b, (short)0, c, false, false);
}

// Async global->LDS 16B copy (per-lane), tracked by ASYNCcnt.
// Low 32 bits of a flat LDS pointer are the LDS byte offset (ISA 10.2).
__device__ __forceinline__ void async_copy_b128(void* lds_dst, const void* gsrc) {
  unsigned loff = (unsigned)(uintptr_t)lds_dst;
  asm volatile("global_load_async_to_lds_b128 %0, %1, off"
               :: "v"(loff), "v"(gsrc) : "memory");
}
__device__ __forceinline__ void async_wait0() {
  asm volatile("s_wait_asynccnt 0x0" ::: "memory");
}

// ---------------------------------------------------------------------------
// GEMM: Y[M,3072] = X[M,3072] @ W[3072,3072] + bias.  128x128 tile, BK=32.
// 8 waves: wave (wm,wn) in 4x2 grid owns a 32x64 subtile = 2x4 WMMA frags.
// ---------------------------------------------------------------------------
__global__ __launch_bounds__(256) void gemm_bias_128x128(
    const float* __restrict__ X, const float* __restrict__ W,
    const float* __restrict__ bias, float* __restrict__ Y)
{
  constexpr int N = 3072, K = 3072;
  constexpr int LDA = 40, LDB = 40;                 // padded bf16 strides
  __shared__ __bf16 As[128 * LDA];                  // As[m][k]
  __shared__ __bf16 Bs[128 * LDB];                  // Bs[n][k]  (transposed)

  const int tid  = threadIdx.x;
  const int lane = tid & 31, wid = tid >> 5;
  const int wm = wid >> 1, wn = wid & 1;
  const int gm0 = blockIdx.y * 128, gn0 = blockIdx.x * 128;
  const int l15 = lane & 15, ksel = lane >> 4;

  f32x8 acc[2][4] = {};

  const int arow = tid >> 1,  acol = (tid & 1) * 16;   // A: 2 thr/row, 16 f32 each
  const int brow = tid >> 3,  bcol = (tid & 7) * 16;   // B: 8 thr/row, 16 f32 each

  for (int k0 = 0; k0 < K; k0 += 32) {
    // ---- stage A tile (f32 -> bf16, packed converts) ----
    {
      const float4* ap = (const float4*)(X + (size_t)(gm0 + arow) * K + k0 + acol);
      float4 t0 = ap[0], t1 = ap[1], t2 = ap[2], t3 = ap[3];
      bf16x8 p0, p1;
      p0[0]=f2bf(t0.x); p0[1]=f2bf(t0.y); p0[2]=f2bf(t0.z); p0[3]=f2bf(t0.w);
      p0[4]=f2bf(t1.x); p0[5]=f2bf(t1.y); p0[6]=f2bf(t1.z); p0[7]=f2bf(t1.w);
      p1[0]=f2bf(t2.x); p1[1]=f2bf(t2.y); p1[2]=f2bf(t2.z); p1[3]=f2bf(t2.w);
      p1[4]=f2bf(t3.x); p1[5]=f2bf(t3.y); p1[6]=f2bf(t3.z); p1[7]=f2bf(t3.w);
      *(bf16x8*)&As[arow * LDA + acol]     = p0;
      *(bf16x8*)&As[arow * LDA + acol + 8] = p1;
    }
    // ---- stage B tile transposed (f32 -> bf16) ----
    {
      const float4* bp = (const float4*)(W + (size_t)(k0 + brow) * N + gn0 + bcol);
      float4 t0 = bp[0], t1 = bp[1], t2 = bp[2], t3 = bp[3];
      float bv[16] = { t0.x,t0.y,t0.z,t0.w, t1.x,t1.y,t1.z,t1.w,
                       t2.x,t2.y,t2.z,t2.w, t3.x,t3.y,t3.z,t3.w };
      #pragma unroll
      for (int j = 0; j < 16; ++j)
        Bs[(bcol + j) * LDB + brow] = f2bf(bv[j]);
    }
    if (k0 + 32 < K) {
      __builtin_prefetch(X + (size_t)(gm0 + arow) * K + k0 + 32 + acol, 0, 1);
      __builtin_prefetch(W + (size_t)(k0 + 32 + brow) * N + gn0 + bcol, 0, 1);
    }
    __syncthreads();

    // ---- fragments per ISA layouts ----
    bf16x16 af[2], bf[4];
    #pragma unroll
    for (int mi = 0; mi < 2; ++mi) {
      const __bf16* p = &As[(wm * 32 + mi * 16 + l15) * LDA];
      bf16x8 lo = *(const bf16x8*)(p + 8 * ksel);        // K: 8k..8k+7
      bf16x8 hi = *(const bf16x8*)(p + 16 + 8 * ksel);   // K: 16+8k..
      af[mi] = SHUF16(lo, hi);
    }
    #pragma unroll
    for (int ni = 0; ni < 4; ++ni) {
      const __bf16* p = &Bs[(wn * 64 + ni * 16 + l15) * LDB + 8 * ksel];
      bf16x8 lo = *(const bf16x8*)p;                     // K: 8k..8k+7
      bf16x8 hi = *(const bf16x8*)(p + 16);
      bf[ni] = SHUF16(lo, hi);
    }
    #pragma unroll
    for (int mi = 0; mi < 2; ++mi)
      #pragma unroll
      for (int ni = 0; ni < 4; ++ni)
        acc[mi][ni] = wmma_bf16(af[mi], bf[ni], acc[mi][ni]);
    __syncthreads();
  }

  // ---- epilogue: add bias, store D-layout (VGPR r -> row r / r+8) ----
  const int mbase = ksel * 8;
  #pragma unroll
  for (int mi = 0; mi < 2; ++mi)
    #pragma unroll
    for (int ni = 0; ni < 4; ++ni) {
      const int gn = gn0 + wn * 64 + ni * 16 + l15;
      const float b = bias[gn];
      #pragma unroll
      for (int r = 0; r < 8; ++r) {
        const int gm = gm0 + wm * 32 + mi * 16 + mbase + r;
        Y[(size_t)gm * N + gn] = acc[mi][ni][r] + b;
      }
    }
}

// ---------------------------------------------------------------------------
// Per-head RMSNorm (eps=1e-6) + interleaved RoPE + f32->bf16.  One wave per
// (token, head); lane owns 4 consecutive dims (= 2 RoPE pairs).
// mode=1: rms+rope (Q/K).  mode=0: convert only (V).
// ---------------------------------------------------------------------------
__global__ __launch_bounds__(256) void rmsrope_cvt(
    const float* __restrict__ src, const float* __restrict__ nw,
    const float* __restrict__ cosb, const float* __restrict__ sinb,
    __bf16* __restrict__ dst, int seq0, int mode)
{
  const int tok  = blockIdx.x;
  const int head = blockIdx.y * 8 + (threadIdx.x >> 5);
  const int lane = threadIdx.x & 31;
  const int d0   = lane * 4;

  float4 x = *(const float4*)(src + (size_t)tok * 3072 + head * 128 + d0);

  if (mode) {
    float ss = x.x*x.x + x.y*x.y + x.z*x.z + x.w*x.w;
    #pragma unroll
    for (int off = 1; off < 32; off <<= 1) ss += __shfl_xor(ss, off, 32);
    const float inv = rsqrtf(ss * (1.0f / 128.0f) + 1e-6f);
    const float4 w = *(const float4*)(nw + d0);
    x.x *= inv * w.x; x.y *= inv * w.y; x.z *= inv * w.z; x.w *= inv * w.w;
    const size_t ro = (size_t)(seq0 + tok) * 128 + d0;
    const float4 c = *(const float4*)(cosb + ro);
    const float4 s = *(const float4*)(sinb + ro);
    float o0 = x.x * c.x - x.y * s.x;     // out[2i]   = x0*cos - x1*sin
    float o1 = x.y * c.y + x.x * s.y;     // out[2i+1] = x1*cos + x0*sin
    float o2 = x.z * c.z - x.w * s.z;
    float o3 = x.w * c.w + x.z * s.w;
    x.x = o0; x.y = o1; x.z = o2; x.w = o3;
  }
  bf16x4 o; o[0]=f2bf(x.x); o[1]=f2bf(x.y); o[2]=f2bf(x.z); o[3]=f2bf(x.w);
  *(bf16x4*)(dst + (size_t)(seq0 + tok) * 3072 + head * 128 + d0) = o;
}

// ---------------------------------------------------------------------------
// Flash attention, D=128.  8 waves/block, 16 q-rows/wave (128 q-rows/block),
// KV chunk = 32.  QK^T and P.V via WMMA; online softmax in f32.
// K chunk staged via async global->LDS; V staged via VGPRs (transposed).
// ---------------------------------------------------------------------------
__global__ __launch_bounds__(256) void flash_attn(
    const __bf16* __restrict__ Qb, const __bf16* __restrict__ Kb,
    const __bf16* __restrict__ Vb, float* __restrict__ Out,
    int q_start, int kv_start, int kv_len)
{
  constexpr int HD  = 3072;
  constexpr int LKS = 136, LVT = 40, LPS = 40;
  __shared__ __bf16 Ks[32 * LKS];        // Ks[tok][d]  (== B-frag layout for QK^T)
  __shared__ __bf16 Vt[128 * LVT];       // Vt[d][tok]  (B-frag layout for P.V)
  __shared__ __bf16 Ps[8][16 * LPS];     // wave-private P staging (16x32)

  const int tid = threadIdx.x, lane = tid & 31, wid = tid >> 5;
  const int head  = blockIdx.y;
  const int qrow0 = q_start + blockIdx.x * 128 + wid * 16;
  const int l15 = lane & 15, ksel = lane >> 4;
  const float scale = 0.08838834764831845f;   // 1/sqrt(128)

  // Q fragments (A layout, 16x32 per frag, 4 frags cover D=128)
  bf16x16 qf[4];
  {
    const __bf16* qp = Qb + (size_t)(qrow0 + l15) * HD + head * 128;
    #pragma unroll
    for (int f = 0; f < 4; ++f) {
      bf16x8 lo = *(const bf16x8*)(qp + f * 32 + 8 * ksel);
      bf16x8 hi = *(const bf16x8*)(qp + f * 32 + 16 + 8 * ksel);
      qf[f] = SHUF16(lo, hi);
    }
  }

  f32x8 o[8] = {};
  float mrow[8], lrow[8];
  #pragma unroll
  for (int r = 0; r < 8; ++r) { mrow[r] = -3.0e38f; lrow[r] = 0.0f; }

  const int ldtok = tid >> 3, lddb = (tid & 7) * 16;   // coop KV loaders

  for (int kc = 0; kc < kv_len; kc += 32) {
    // ---- stage K chunk (async DMA to LDS, no VGPR round-trip) ----
    {
      const size_t row = (size_t)(kv_start + kc + ldtok) * HD + head * 128 + lddb;
      async_copy_b128(&Ks[ldtok * LKS + lddb],     Kb + row);
      async_copy_b128(&Ks[ldtok * LKS + lddb + 8], Kb + row + 8);
      // ---- stage V chunk transposed via VGPRs ----
      bf16x8 v0 = *(const bf16x8*)(Vb + row);
      bf16x8 v1 = *(const bf16x8*)(Vb + row + 8);
      #pragma unroll
      for (int j = 0; j < 8; ++j) Vt[(lddb + j)     * LVT + ldtok] = v0[j];
      #pragma unroll
      for (int j = 0; j < 8; ++j) Vt[(lddb + 8 + j) * LVT + ldtok] = v1[j];
    }
    async_wait0();
    __syncthreads();

    // ---- S = scale * Q K^T   (two 16x16 tiles over 32 kv cols) ----
    f32x8 s[2] = {};
    #pragma unroll
    for (int t = 0; t < 2; ++t)
      #pragma unroll
      for (int f = 0; f < 4; ++f) {
        const __bf16* kp = &Ks[(t * 16 + l15) * LKS + f * 32 + 8 * ksel];
        bf16x8 lo = *(const bf16x8*)kp;
        bf16x8 hi = *(const bf16x8*)(kp + 16);
        s[t] = wmma_bf16(qf[f], SHUF16(lo, hi), s[t]);
      }
    #pragma unroll
    for (int t = 0; t < 2; ++t)
      #pragma unroll
      for (int r = 0; r < 8; ++r) s[t][r] *= scale;

    // ---- online softmax (xor masks 1,2,4,8 keep lane-halves = row groups) ----
    float alpha[8];
    #pragma unroll
    for (int r = 0; r < 8; ++r) {
      float v = fmaxf(s[0][r], s[1][r]);
      #pragma unroll
      for (int off = 1; off < 16; off <<= 1) v = fmaxf(v, __shfl_xor(v, off, 32));
      const float nm = fmaxf(mrow[r], v);
      alpha[r] = __expf(mrow[r] - nm);
      mrow[r] = nm;
    }
    #pragma unroll
    for (int r = 0; r < 8; ++r) {
      s[0][r] = __expf(s[0][r] - mrow[r]);
      s[1][r] = __expf(s[1][r] - mrow[r]);
      float v = s[0][r] + s[1][r];
      #pragma unroll
      for (int off = 1; off < 16; off <<= 1) v += __shfl_xor(v, off, 32);
      lrow[r] = lrow[r] * alpha[r] + v;
    }
    #pragma unroll
    for (int f = 0; f < 8; ++f)
      #pragma unroll
      for (int r = 0; r < 8; ++r) o[f][r] *= alpha[r];

    // ---- P: D-layout -> A-layout via wave-private LDS (DS ops in-order) ----
    __bf16* ps = Ps[wid];
    #pragma unroll
    for (int t = 0; t < 2; ++t)
      #pragma unroll
      for (int r = 0; r < 8; ++r)
        ps[(ksel * 8 + r) * LPS + t * 16 + l15] = f2bf(s[t][r]);
    bf16x16 pf;
    {
      const __bf16* pp = &ps[l15 * LPS + 8 * ksel];
      bf16x8 lo = *(const bf16x8*)pp;
      bf16x8 hi = *(const bf16x8*)(pp + 16);
      pf = SHUF16(lo, hi);
    }

    // ---- O += P.V ----
    #pragma unroll
    for (int f = 0; f < 8; ++f) {
      const __bf16* vp = &Vt[(f * 16 + l15) * LVT + 8 * ksel];
      bf16x8 lo = *(const bf16x8*)vp;
      bf16x8 hi = *(const bf16x8*)(vp + 16);
      o[f] = wmma_bf16(pf, SHUF16(lo, hi), o[f]);
    }
    __syncthreads();
  }

  // ---- epilogue: O /= l, store ----
  float inv[8];
  #pragma unroll
  for (int r = 0; r < 8; ++r) inv[r] = 1.0f / lrow[r];
  const int mbase = ksel * 8;
  #pragma unroll
  for (int f = 0; f < 8; ++f) {
    const int gn = head * 128 + f * 16 + l15;
    #pragma unroll
    for (int r = 0; r < 8; ++r)
      Out[(size_t)(qrow0 + mbase + r) * HD + gn] = o[f][r] * inv[r];
  }
}

// ---------------------------------------------------------------------------
// Host orchestration
// ---------------------------------------------------------------------------
extern "C" void kernel_launch(void* const* d_in, const int* in_sizes, int n_in,
                              void* d_out, int out_size, void* d_ws, size_t ws_size,
                              hipStream_t stream) {
  constexpr int TXT = 512, S = 2048, STOT = 2560, HD = 3072;
  (void)in_sizes; (void)n_in; (void)out_size; (void)ws_size;

  const float* hs   = (const float*)d_in[0];
  const float* enc  = (const float*)d_in[1];
  const float* cosb = (const float*)d_in[2];
  const float* sinb = (const float*)d_in[3];
  const float* wq  = (const float*)d_in[4];  const float* bq  = (const float*)d_in[5];
  const float* wk  = (const float*)d_in[6];  const float* bk  = (const float*)d_in[7];
  const float* wv  = (const float*)d_in[8];  const float* bv  = (const float*)d_in[9];
  const float* waq = (const float*)d_in[10]; const float* baq = (const float*)d_in[11];
  const float* wak = (const float*)d_in[12]; const float* bak = (const float*)d_in[13];
  const float* wav = (const float*)d_in[14]; const float* bav = (const float*)d_in[15];
  const float* nq  = (const float*)d_in[16]; const float* nk  = (const float*)d_in[17];
  const float* naq = (const float*)d_in[18]; const float* nak = (const float*)d_in[19];
  const float* wo  = (const float*)d_in[20]; const float* bo  = (const float*)d_in[21];
  const float* wao = (const float*)d_in[22]; const float* bao = (const float*)d_in[23];
  // d_in[24] = num_ref_tokens (device scalar) == REF == 512, fixed by problem shape.
  float* out = (float*)d_out;

  const size_t SEQELTS = (size_t)STOT * HD;
  float*  tmp = (float*)d_ws;                               // STOT*3072 f32 (reused)
  __bf16* Qb  = (__bf16*)((char*)d_ws + SEQELTS * 4);
  __bf16* Kb  = Qb + SEQELTS;
  __bf16* Vb  = Kb + SEQELTS;

  const dim3 blk(256);
  auto gemm = [&](const float* X, const float* W, const float* b, float* Y, int M) {
    gemm_bias_128x128<<<dim3(HD / 128, M / 128), blk, 0, stream>>>(X, W, b, Y);
  };
  auto rr = [&](const float* src, const float* w, __bf16* dst, int rows, int seq0, int mode) {
    rmsrope_cvt<<<dim3(rows, 3), blk, 0, stream>>>(src, w, cosb, sinb, dst, seq0, mode);
  };

  // QKV for hidden stream (seq positions TXT..STOT) and encoder (0..TXT)
  gemm(hs,  wq,  bq,  tmp, S);   rr(tmp, nq,      Qb, S,   TXT, 1);
  gemm(hs,  wk,  bk,  tmp, S);   rr(tmp, nk,      Kb, S,   TXT, 1);
  gemm(hs,  wv,  bv,  tmp, S);   rr(tmp, nullptr, Vb, S,   TXT, 0);
  gemm(enc, waq, baq, tmp, TXT); rr(tmp, naq,     Qb, TXT, 0,   1);
  gemm(enc, wak, bak, tmp, TXT); rr(tmp, nak,     Kb, TXT, 0,   1);
  gemm(enc, wav, bav, tmp, TXT); rr(tmp, nullptr, Vb, TXT, 0,   0);

  // Structured attention into tmp (reused): txt->all, img->all, ref->ref
  flash_attn<<<dim3(TXT / 128, 24),  blk, 0, stream>>>(Qb, Kb, Vb, tmp, 0,    0,   STOT);
  flash_attn<<<dim3(1536 / 128, 24), blk, 0, stream>>>(Qb, Kb, Vb, tmp, 1024, 0,   STOT);
  flash_attn<<<dim3(TXT / 128, 24),  blk, 0, stream>>>(Qb, Kb, Vb, tmp, TXT,  TXT, TXT);

  // Output projections: hs_out = attn[TXT:] @ wo + bo ; enc_out = attn[:TXT] @ wao + bao
  gemm(tmp + (size_t)TXT * HD, wo,  bo,  out,                    S);
  gemm(tmp,                    wao, bao, out + (size_t)S * HD,   TXT);
}
